// VanillaRNN_83073257439278
// MI455X (gfx1250) — matrix operand order repaired and checked by reading, not compiled, not run on hardware
//
#include <hip/hip_runtime.h>
#include <math.h>

typedef _Float16 half8  __attribute__((ext_vector_type(8)));
typedef _Float16 half16 __attribute__((ext_vector_type(16)));
typedef float    float8 __attribute__((ext_vector_type(8)));
typedef float    f32x4  __attribute__((ext_vector_type(4)));

#define BATCH 512
#define SEQL  512
#define INDIM 128
#define HID   256
#define NCLS  10
#define ROWS  16            // batch rows per workgroup
#define HPAD  (HID + 8)     // padded f16 row (528B = +4 banks/row -> conflict-free A loads)
#define IPAD  (INDIM + 8)   // padded f16 row for WhxT
#define XPADH (INDIM + 8)   // padded f16 row for staged x (272B = +4 banks/row)

#if __has_builtin(__builtin_amdgcn_tanhf)
#define DEV_TANH(v) __builtin_amdgcn_tanhf(v)   // native v_tanh_f32 on gfx1250
#else
#define DEV_TANH(v) tanhf(v)
#endif

static __device__ __forceinline__ half16 cat8(half8 lo, half8 hi) {
  return __builtin_shufflevector(lo, hi, 0, 1, 2, 3, 4, 5, 6, 7,
                                 8, 9, 10, 11, 12, 13, 14, 15);
}

__global__ __launch_bounds__(256, 1)
void rnn_fused_wmma(const float* __restrict__ x,   const float* __restrict__ Whx,
                    const float* __restrict__ Whh, const float* __restrict__ Wph,
                    const float* __restrict__ bh,  const float* __restrict__ bp,
                    float* __restrict__ out)
{
  __shared__ __align__(16) _Float16 sWhhT[HID][HPAD];    // [n][k]  132 KB
  __shared__ __align__(16) _Float16 sWhxT[HID][IPAD];    // [n][k]   68 KB
  __shared__ __align__(16) _Float16 sH[2][ROWS][HPAD];   // h double buffer, 16.5 KB
  __shared__ __align__(16) _Float16 sX[2][ROWS][XPADH];  // x_t double buffer (f16), 8.5 KB
  __shared__ float                  sBh[HID];

  const int tid   = threadIdx.x;
  const int lane  = tid & 31;        // wave32
  const int wave  = tid >> 5;        // 0..7
  const int bBase = blockIdx.x * ROWS;
  const int mrow  = lane & 15;       // M (for A/C) or N (for B) within tile
  const int hiH   = lane >> 4;       // lane half selector
  const int n0    = wave * 32;       // this wave owns N columns [n0, n0+32) (two 16-wide tiles)

  // ---- one-time: transpose+convert weights to f16 in LDS (tid == n, coalesced reads)
  for (int k = 0; k < HID; ++k)
    sWhhT[tid][k] = (_Float16)Whh[k * HID + tid];
  for (int k = 0; k < INDIM; ++k)
    sWhxT[tid][k] = (_Float16)Whx[k * HID + tid];
  sBh[tid] = bh[tid];
  for (int i = tid; i < ROWS * HPAD; i += 256)        // h0 = 0
    (&sH[0][0][0])[i] = (_Float16)0.0f;

  // stage x_0 as f16: thread covers row xm, 8 consecutive elements
  const int xm = tid >> 4;
  const int xk = (tid & 15) * 8;
  {
    const float* src = x + ((size_t)(bBase + xm) * SEQL + 0) * INDIM + xk;
    f32x4 p0 = *(const f32x4*)src;
    f32x4 p1 = *(const f32x4*)(src + 4);
    half8 hx;
#pragma unroll
    for (int j = 0; j < 4; ++j) { hx[j] = (_Float16)p0[j]; hx[4 + j] = (_Float16)p1[j]; }
    *(half8*)&sX[0][xm][xk] = hx;
  }
  __syncthreads();

  // ---- hoist all B fragments (weights) into registers: constant over time
  // B k order used here (contiguous, same as A): lane l: N = l%16, K = kBase + (l<16 ? 0..15 : 16..31)
  half16 bWhh[2][8];
  half16 bWhx[2][4];
#pragma unroll
  for (int nt = 0; nt < 2; ++nt) {
#pragma unroll
    for (int kt = 0; kt < 8; ++kt) {
      const _Float16* p = &sWhhT[n0 + nt * 16 + mrow][kt * 32 + hiH * 16];
      bWhh[nt][kt] = cat8(*(const half8*)p, *(const half8*)(p + 8));
    }
#pragma unroll
    for (int kt = 0; kt < 4; ++kt) {
      const _Float16* p = &sWhxT[n0 + nt * 16 + mrow][kt * 32 + hiH * 16];
      bWhx[nt][kt] = cat8(*(const half8*)p, *(const half8*)(p + 8));
    }
  }
  const float bv0 = sBh[n0 + mrow];
  const float bv1 = sBh[n0 + 16 + mrow];

  // ---- recurrence: 512 sequential steps, one barrier per step
  for (int t = 0; t < SEQL; ++t) {
    const int cur = t & 1, nxt = cur ^ 1;

    // prefetch x_{t+1}; global-load latency hidden under the WMMA work below
    half8 hx = {};
    const bool pf = (t + 1 < SEQL);
    if (pf) {
      const float* src = x + ((size_t)(bBase + xm) * SEQL + (t + 1)) * INDIM + xk;
      f32x4 p0 = *(const f32x4*)src;
      f32x4 p1 = *(const f32x4*)(src + 4);
#pragma unroll
      for (int j = 0; j < 4; ++j) { hx[j] = (_Float16)p0[j]; hx[4 + j] = (_Float16)p1[j]; }
    }

    float8 acc0, acc1;
#pragma unroll
    for (int r = 0; r < 8; ++r) { acc0[r] = bv0; acc1[r] = bv1; }

    // x_t @ Whx : K = 128 (4 tiles of 32), A f16 from LDS
    // A layout: lane half h holds K+16h..16h+15 (contiguous, matches B)
#pragma unroll
    for (int kt = 0; kt < 4; ++kt) {
      const _Float16* pa = &sX[cur][mrow][kt * 32 + hiH * 16];
      half16 a = cat8(*(const half8*)pa, *(const half8*)(pa + 8));
      acc0 = __builtin_amdgcn_wmma_f32_16x16x32_f16(false, a, false, bWhx[0][kt],
                                                    (short)0, acc0, false, false);
      acc1 = __builtin_amdgcn_wmma_f32_16x16x32_f16(false, a, false, bWhx[1][kt],
                                                    (short)0, acc1, false, false);
    }

    // h_t @ Whh : K = 256 (8 tiles of 32)
#pragma unroll
    for (int kt = 0; kt < 8; ++kt) {
      const _Float16* pa = &sH[cur][mrow][kt * 32 + hiH * 16];
      half16 a = cat8(*(const half8*)pa, *(const half8*)(pa + 8));
      acc0 = __builtin_amdgcn_wmma_f32_16x16x32_f16(false, a, false, bWhh[0][kt],
                                                    (short)0, acc0, false, false);
      acc1 = __builtin_amdgcn_wmma_f32_16x16x32_f16(false, a, false, bWhh[1][kt],
                                                    (short)0, acc1, false, false);
    }

    // commit x_{t+1} stage (single 16B LDS store per thread)
    if (pf) *(half8*)&sX[nxt][xm][xk] = hx;

    // h_{t+1} = tanh(acc); C/D layout: VGPR r, lanes<16 -> M=r, lanes>=16 -> M=r+8; N=lane%16
#pragma unroll
    for (int r = 0; r < 8; ++r) {
      const int mm = r + hiH * 8;
      sH[nxt][mm][n0 + mrow]      = (_Float16)DEV_TANH(acc0[r]);
      sH[nxt][mm][n0 + 16 + mrow] = (_Float16)DEV_TANH(acc1[r]);
    }
    __syncthreads();
  }

  // ---- classifier: out = hT @ Wph + bp  (16 x 10 per WG, one thread per output)
  if (tid < ROWS * NCLS) {
    const int m = tid / NCLS, n = tid % NCLS;
    float acc = bp[n];
    const _Float16* hrow = sH[SEQL & 1][m];
    for (int k = 0; k < HID; ++k)
      acc += (float)hrow[k] * Wph[k * NCLS + n];
    out[(size_t)(bBase + m) * NCLS + n] = acc;
  }
}

extern "C" void kernel_launch(void* const* d_in, const int* in_sizes, int n_in,
                              void* d_out, int out_size, void* d_ws, size_t ws_size,
                              hipStream_t stream) {
  (void)in_sizes; (void)n_in; (void)d_ws; (void)ws_size; (void)out_size;
  const float* x   = (const float*)d_in[0];
  const float* Whx = (const float*)d_in[1];
  const float* Whh = (const float*)d_in[2];
  const float* Wph = (const float*)d_in[3];
  const float* bh  = (const float*)d_in[4];
  const float* bp  = (const float*)d_in[5];
  rnn_fused_wmma<<<BATCH / ROWS, 256, 0, stream>>>(x, Whx, Whh, Wph, bh, bp,
                                                   (float*)d_out);
}
